// MA_module_with_LSTM_out_7121055776924
// MI455X (gfx1250) — matrix operand order repaired and checked by reading, not compiled, not run on hardware
//
#include <hip/hip_runtime.h>
#include <hip/hip_bf16.h>

// ---------------- problem constants ----------------
#define T_   4096
#define IN_  1024
#define E_   512
#define G4_  2048   // 4*E

typedef __attribute__((ext_vector_type(16))) __bf16 v16bf;
typedef __attribute__((ext_vector_type(8)))  __bf16 v8bf;
typedef __attribute__((ext_vector_type(8)))  float  v8f;
typedef __attribute__((ext_vector_type(4)))  int    v4i;

// ---- CDNA5 async global->LDS copy (ASYNCcnt path), feature-gated ----
#if defined(__AMDGCN__) && __has_builtin(__builtin_amdgcn_global_load_async_to_lds_b128)
#define HAS_ASYNC_LDS 1
#else
#define HAS_ASYNC_LDS 0
#endif

#if HAS_ASYNC_LDS
typedef __attribute__((address_space(1))) v4i gv4i;   // global 128-bit chunk
typedef __attribute__((address_space(3))) v4i lv4i;   // LDS 128-bit chunk
__device__ __forceinline__ void async_copy_b128(const void* g, void* l) {
  __builtin_amdgcn_global_load_async_to_lds_b128((gv4i*)g, (lv4i*)l, 0, 0);
}
__device__ __forceinline__ void async_wait0() {
#if __has_builtin(__builtin_amdgcn_s_wait_asynccnt)
  __builtin_amdgcn_s_wait_asynccnt(0);
#else
  asm volatile("s_wait_asynccnt 0x0" ::: "memory");
#endif
}
#endif

// ================= conversion / layout kernels =================

__global__ void k_f2bf(const float* __restrict__ s, __bf16* __restrict__ d, long n) {
  long i = (long)blockIdx.x * blockDim.x + threadIdx.x;
  long stride = (long)gridDim.x * blockDim.x;
  for (; i < n; i += stride) d[i] = (__bf16)s[i];
}

// src [rows x cols] f32 row-major -> dst [cols x rows] bf16 row-major
__global__ void k_f2bf_t(const float* __restrict__ s, __bf16* __restrict__ d,
                         int rows, int cols) {
  long n = (long)rows * cols;
  long i = (long)blockIdx.x * blockDim.x + threadIdx.x;
  long stride = (long)gridDim.x * blockDim.x;
  for (; i < n; i += stride) {
    int r = (int)(i / cols), c = (int)(i % cols);
    d[(long)c * rows + r] = (__bf16)s[i];
  }
}

// bf16 transpose: src [rows x cols] -> dst [cols x rows]
__global__ void k_bf_t(const __bf16* __restrict__ s, __bf16* __restrict__ d,
                       int rows, int cols) {
  long n = (long)rows * cols;
  long i = (long)blockIdx.x * blockDim.x + threadIdx.x;
  long stride = (long)gridDim.x * blockDim.x;
  for (; i < n; i += stride) {
    int r = (int)(i / cols), c = (int)(i % cols);
    d[(long)c * rows + r] = s[i];
  }
}

__global__ void k_bias_sum(const float* __restrict__ a, const float* __restrict__ b,
                           float* __restrict__ d, int n) {
  int i = blockIdx.x * blockDim.x + threadIdx.x;
  if (i < n) d[i] = a[i] + b[i];
}

// ================= WMMA fragment helpers (ISA 7.12.2 layouts) ==========
// A (16x32 bf16): lane<16 -> row=lane, K[0..7]+[16..23]; lane>=16 -> row=lane-16, K[8..15]+[24..31]
__device__ __forceinline__ v16bf load_a_frag(const __bf16* base, long ld, int row, int k0, int lane) {
  int hi = lane >> 4;
  const __bf16* p = base + (long)row * ld + k0 + hi * 8;
  v8bf lo = *(const v8bf*)p;
  v8bf hh = *(const v8bf*)(p + 16);
  v16bf r;
#pragma unroll
  for (int i = 0; i < 8; ++i) { r[i] = lo[i]; r[8 + i] = hh[i]; }
  return r;
}
// B (32x16 bf16) taken from row-major W[N x K]: lane<16 -> col=lane, K[0..7]+[16..23]; lane>=16 -> col=lane-16, K[8..15]+[24..31]
__device__ __forceinline__ v16bf load_b_frag(const __bf16* base, long ld, int col, int k0, int lane) {
  int hi = lane >> 4;
  const __bf16* p = base + (long)col * ld + k0 + hi * 8;
  v8bf lo = *(const v8bf*)p;
  v8bf hh = *(const v8bf*)(p + 16);
  v16bf r;
#pragma unroll
  for (int i = 0; i < 8; ++i) { r[i] = lo[i]; r[8 + i] = hh[i]; }
  return r;
}

// ================= NT GEMM:  C[MxN] = X[MxK] * W[NxK]^T (+bias) ========
// block = 128 threads (4 waves); wave tile 16x64; block tile 16x256.
// X tile (16 x K) is staged once into LDS via async global->LDS copies and
// shared by all 4 waves (A-frags come from ds_load_b128, conflict-free with
// the K+8 bf16 row-stride padding: stride = (K+8)/2 dwords == 4 mod 64 banks).
__global__ void k_gemm_nt(const __bf16* __restrict__ X, const __bf16* __restrict__ W,
                          float* __restrict__ Cf, __bf16* __restrict__ Cb,
                          const float* __restrict__ bias, int M, int N, int K) {
  __shared__ __bf16 sX[16 * (IN_ + 8)];   // sized for max K = 1024
  const int ldsK = K + 8;

  const int lane = threadIdx.x & 31;
  const int wave = threadIdx.x >> 5;
  const int l15  = lane & 15;
  const int hi   = lane >> 4;
  const int m0 = blockIdx.y * 16;
  const int n0 = blockIdx.x * 256 + wave * 64;

  // ---- stage X[m0..m0+15, 0..K) into LDS (16B chunks, all threads) ----
  {
    const int chunksPerRow = K >> 3;            // 8 bf16 per b128
    const int nChunks = 16 * chunksPerRow;      // divisible by 128
    for (int c = threadIdx.x; c < nChunks; c += 128) {
      int row  = c / chunksPerRow;
      int col8 = c - row * chunksPerRow;
      const __bf16* g = X + (long)(m0 + row) * K + col8 * 8;
      __bf16* l = &sX[row * ldsK + col8 * 8];
#if HAS_ASYNC_LDS
      async_copy_b128(g, l);
#else
      *(v8bf*)l = *(const v8bf*)g;
#endif
    }
#if HAS_ASYNC_LDS
    async_wait0();
#endif
    __syncthreads();
  }

  v8f acc[4] = {};
  for (int k0 = 0; k0 < K; k0 += 32) {
    // A-frag from LDS
    const __bf16* pa = &sX[l15 * ldsK + k0 + hi * 8];
    v8bf lo = *(const v8bf*)pa;
    v8bf hh = *(const v8bf*)(pa + 16);
    v16bf a;
#pragma unroll
    for (int i = 0; i < 8; ++i) { a[i] = lo[i]; a[8 + i] = hh[i]; }
#pragma unroll
    for (int t = 0; t < 4; ++t) {
      v16bf b = load_b_frag(W, K, n0 + t * 16 + l15, k0, lane);
      acc[t] = __builtin_amdgcn_wmma_f32_16x16x32_bf16(false, a, false, b,
                                                       (short)0, acc[t], false, false);
    }
  }
  const int rbase = m0 + hi * 8;
#pragma unroll
  for (int t = 0; t < 4; ++t) {
    int col = n0 + t * 16 + l15;
    float bb = bias ? bias[col] : 0.0f;
#pragma unroll
    for (int i = 0; i < 8; ++i) {
      float v = acc[t][i] + bb;
      long idx = (long)(rbase + i) * N + col;
      if (Cf) Cf[idx] = v;
      if (Cb) Cb[idx] = (__bf16)v;
    }
  }
}

// ============ flash attention + u_mod gate (16 queries / block) ==========
// scores = u a^T (unscaled), softmax over keys, out = P b ; umod = u * out
// block = 128 threads (4 waves); wave owns 128 E-columns of O.
__global__ void k_flash(const __bf16* __restrict__ u_bf, const __bf16* __restrict__ a_bf,
                        const __bf16* __restrict__ bT_bf, __bf16* __restrict__ umod_bf) {
  __shared__ float  sPart[4][16 * 32];
  __shared__ float  sS[16 * 32];
  __shared__ __bf16 sP[16 * 32];
  __shared__ float  sM[16], sL[16], sScale[16];

  const int lane = threadIdx.x & 31;
  const int wave = threadIdx.x >> 5;
  const int l15  = lane & 15;
  const int hi   = lane >> 4;
  const int q0   = blockIdx.x * 16;
  const int e_w  = wave * 128;

  v8f O[8] = {};
  if (threadIdx.x < 16) { sM[threadIdx.x] = -1e30f; sL[threadIdx.x] = 0.0f; }
  __syncthreads();

  for (int kb = 0; kb < T_; kb += 32) {
    // --- partial S (16 q x 32 keys) over this wave's 128-wide E slice ---
    v8f s0 = {}, s1 = {};
    for (int e0 = e_w; e0 < e_w + 128; e0 += 32) {
      v16bf a  = load_a_frag(u_bf, E_, q0 + l15, e0, lane);
      v16bf b0 = load_b_frag(a_bf, E_, kb + l15, e0, lane);        // keys kb..kb+15
      v16bf b1 = load_b_frag(a_bf, E_, kb + 16 + l15, e0, lane);   // keys kb+16..kb+31
      s0 = __builtin_amdgcn_wmma_f32_16x16x32_bf16(false, a, false, b0, (short)0, s0, false, false);
      s1 = __builtin_amdgcn_wmma_f32_16x16x32_bf16(false, a, false, b1, (short)0, s1, false, false);
    }
    {
      float* dst = sPart[wave];
#pragma unroll
      for (int i = 0; i < 8; ++i) {
        dst[(hi * 8 + i) * 32 + l15]      = s0[i];
        dst[(hi * 8 + i) * 32 + 16 + l15] = s1[i];
      }
    }
    __syncthreads();

    // --- reduce partials + online softmax stats (one thread per query row) ---
    if (threadIdx.x < 16) {
      int r = threadIdx.x;
      float mb = -1e30f;
      for (int c = 0; c < 32; ++c) {
        float v = sPart[0][r * 32 + c] + sPart[1][r * 32 + c] +
                  sPart[2][r * 32 + c] + sPart[3][r * 32 + c];
        sS[r * 32 + c] = v;
        mb = fmaxf(mb, v);
      }
      float m_old = sM[r];
      float m_new = fmaxf(m_old, mb);
      float scale = __expf(m_old - m_new);
      float lsum = 0.0f;
      for (int c = 0; c < 32; ++c) {
        float p = __expf(sS[r * 32 + c] - m_new);
        sP[r * 32 + c] = (__bf16)p;
        lsum += p;
      }
      sL[r] = sL[r] * scale + lsum;
      sM[r] = m_new;
      sScale[r] = scale;
    }
    __syncthreads();

    // --- rescale O and accumulate P @ b (bT is [E x T] row-major) ---
    v16bf pa;
    {
      const __bf16* pp = &sP[l15 * 32 + hi * 8];
#pragma unroll
      for (int i = 0; i < 8; ++i) { pa[i] = pp[i]; pa[8 + i] = pp[16 + i]; }
    }
#pragma unroll
    for (int t = 0; t < 8; ++t) {
      int col = e_w + t * 16 + l15;
      v16bf b = load_b_frag(bT_bf, T_, col, kb, lane);
#pragma unroll
      for (int i = 0; i < 8; ++i) O[t][i] *= sScale[hi * 8 + i];
      O[t] = __builtin_amdgcn_wmma_f32_16x16x32_bf16(false, pa, false, b, (short)0, O[t], false, false);
    }
    __syncthreads();
  }

  // --- epilogue: umod = u * (O / l) ---
#pragma unroll
  for (int t = 0; t < 8; ++t) {
    int col = e_w + t * 16 + l15;
#pragma unroll
    for (int i = 0; i < 8; ++i) {
      int r = q0 + hi * 8 + i;
      float uval = (float)u_bf[(long)r * E_ + col];
      float o = O[t][i] / sL[hi * 8 + i];
      umod_bf[(long)r * E_ + col] = (__bf16)(uval * o);
    }
  }
}

// ============ persistent sequential LSTM (2 blocks: fwd, bwd) ===========
// Gx already holds u_mod@Wih^T + (b_ih+b_hh), shape [T x 2048], torch gate order i,f,g,o
__global__ void k_lstm(const float* __restrict__ Gf, const float* __restrict__ Gb,
                       const __bf16* __restrict__ Whhf, const __bf16* __restrict__ Whhb,
                       float* __restrict__ hf, float* __restrict__ hb) {
  __shared__ float sh[E_];
  __shared__ float sg[G4_];
  const int rev = blockIdx.x;                // 0 = forward, 1 = backward
  const float* __restrict__ Gx  = rev ? Gb : Gf;
  const __bf16* __restrict__ Whh = rev ? Whhb : Whhf;
  float* __restrict__ hout = rev ? hb : hf;

  const int tid = threadIdx.x;               // 1024 threads
  if (tid < E_) sh[tid] = 0.0f;
  float c_reg = 0.0f;
  const __bf16* w0 = Whh + (long)tid * E_;
  const __bf16* w1 = Whh + (long)(tid + 1024) * E_;
  __syncthreads();

  for (int s = 0; s < T_; ++s) {
    const int t = rev ? (T_ - 1 - s) : s;
    float g0 = Gx[(long)t * G4_ + tid];
    float g1 = Gx[(long)t * G4_ + tid + 1024];
    if (s + 1 < T_) {
      const int tn = rev ? (T_ - 2 - s) : (s + 1);
      __builtin_prefetch(&Gx[(long)tn * G4_ + tid], 0, 0);
    }
    for (int e = 0; e < E_; e += 8) {
      v8bf a0 = *(const v8bf*)(w0 + e);
      v8bf a1 = *(const v8bf*)(w1 + e);
#pragma unroll
      for (int i = 0; i < 8; ++i) {
        float h = sh[e + i];
        g0 += (float)a0[i] * h;
        g1 += (float)a1[i] * h;
      }
    }
    sg[tid] = g0;
    sg[tid + 1024] = g1;
    __syncthreads();
    if (tid < E_) {
      float ig = sg[tid], fg = sg[E_ + tid], gg = sg[2 * E_ + tid], og = sg[3 * E_ + tid];
      ig = 1.0f / (1.0f + __expf(-ig));
      fg = 1.0f / (1.0f + __expf(-fg));
      gg = tanhf(gg);
      og = 1.0f / (1.0f + __expf(-og));
      c_reg = fg * c_reg + ig * gg;
      float h = og * tanhf(c_reg);
      sh[tid] = h;
      hout[(long)t * E_ + tid] = h;
    }
    __syncthreads();
  }
}

// ============ final projection: out[t] = [hf|hb] . final_w + b ==========
__global__ void k_final(const float* __restrict__ hf, const float* __restrict__ hb,
                        const float* __restrict__ fw, const float* __restrict__ fb,
                        float* __restrict__ out) {
  __shared__ float red[128];
  const int t = blockIdx.x;
  float s = 0.0f;
  for (int e = threadIdx.x; e < E_; e += 128)
    s += hf[(long)t * E_ + e] * fw[e] + hb[(long)t * E_ + e] * fw[E_ + e];
  red[threadIdx.x] = s;
  __syncthreads();
  for (int off = 64; off > 0; off >>= 1) {
    if (threadIdx.x < off) red[threadIdx.x] += red[threadIdx.x + off];
    __syncthreads();
  }
  if (threadIdx.x == 0) out[t] = red[0] + fb[0];
}

// ========================= host launcher =========================
extern "C" void kernel_launch(void* const* d_in, const int* in_sizes, int n_in,
                              void* d_out, int out_size, void* d_ws, size_t ws_size,
                              hipStream_t stream) {
  const float* x      = (const float*)d_in[0];
  const float* A      = (const float*)d_in[1];
  const float* B      = (const float*)d_in[2];
  const float* U      = (const float*)d_in[3];
  const float* w_ih_f = (const float*)d_in[4];
  const float* w_hh_f = (const float*)d_in[5];
  const float* b_ih_f = (const float*)d_in[6];
  const float* b_hh_f = (const float*)d_in[7];
  const float* w_ih_b = (const float*)d_in[8];
  const float* w_hh_b = (const float*)d_in[9];
  const float* b_ih_b = (const float*)d_in[10];
  const float* b_hh_b = (const float*)d_in[11];
  const float* fin_w  = (const float*)d_in[12];
  const float* fin_b  = (const float*)d_in[13];
  float* out = (float*)d_out;

  char* ws = (char*)d_ws;
  auto alloc = [&](size_t bytes) -> char* {
    char* p = ws;
    ws += (bytes + 255) & ~(size_t)255;
    return p;
  };
  __bf16* xb    = (__bf16*)alloc((size_t)T_ * IN_ * 2);
  __bf16* At    = (__bf16*)alloc((size_t)E_ * IN_ * 2);
  __bf16* Bt    = (__bf16*)alloc((size_t)E_ * IN_ * 2);
  __bf16* Ut    = (__bf16*)alloc((size_t)E_ * IN_ * 2);
  __bf16* u_bf  = (__bf16*)alloc((size_t)T_ * E_ * 2);
  __bf16* a_bf  = (__bf16*)alloc((size_t)T_ * E_ * 2);
  __bf16* b_bf  = (__bf16*)alloc((size_t)T_ * E_ * 2);
  __bf16* bT_bf = (__bf16*)alloc((size_t)E_ * T_ * 2);
  __bf16* um_bf = (__bf16*)alloc((size_t)T_ * E_ * 2);
  __bf16* Wihf  = (__bf16*)alloc((size_t)G4_ * E_ * 2);
  __bf16* Whhf  = (__bf16*)alloc((size_t)G4_ * E_ * 2);
  __bf16* Wihb  = (__bf16*)alloc((size_t)G4_ * E_ * 2);
  __bf16* Whhb  = (__bf16*)alloc((size_t)G4_ * E_ * 2);
  float*  biasf = (float*)alloc((size_t)G4_ * 4);
  float*  biasb = (float*)alloc((size_t)G4_ * 4);
  float*  Gfx   = (float*)alloc((size_t)T_ * G4_ * 4);
  float*  Gbx   = (float*)alloc((size_t)T_ * G4_ * 4);
  float*  hf    = (float*)alloc((size_t)T_ * E_ * 4);
  float*  hb    = (float*)alloc((size_t)T_ * E_ * 4);

  // ---- conversions / transposes ----
  k_f2bf  <<<2048, 256, 0, stream>>>(x, xb, (long)T_ * IN_);
  k_f2bf_t<<<1024, 256, 0, stream>>>(A, At, IN_, E_);
  k_f2bf_t<<<1024, 256, 0, stream>>>(B, Bt, IN_, E_);
  k_f2bf_t<<<1024, 256, 0, stream>>>(U, Ut, IN_, E_);
  k_f2bf  <<<2048, 256, 0, stream>>>(w_ih_f, Wihf, (long)G4_ * E_);
  k_f2bf  <<<2048, 256, 0, stream>>>(w_hh_f, Whhf, (long)G4_ * E_);
  k_f2bf  <<<2048, 256, 0, stream>>>(w_ih_b, Wihb, (long)G4_ * E_);
  k_f2bf  <<<2048, 256, 0, stream>>>(w_hh_b, Whhb, (long)G4_ * E_);
  k_bias_sum<<<G4_ / 256, 256, 0, stream>>>(b_ih_f, b_hh_f, biasf, G4_);
  k_bias_sum<<<G4_ / 256, 256, 0, stream>>>(b_ih_b, b_hh_b, biasb, G4_);

  // ---- projections: u = x U, a = x A, b = x B (bf16 outputs) ----
  dim3 gproj(E_ / 256, T_ / 16);
  k_gemm_nt<<<gproj, 128, 0, stream>>>(xb, Ut, nullptr, u_bf, nullptr, T_, E_, IN_);
  k_gemm_nt<<<gproj, 128, 0, stream>>>(xb, At, nullptr, a_bf, nullptr, T_, E_, IN_);
  k_gemm_nt<<<gproj, 128, 0, stream>>>(xb, Bt, nullptr, b_bf, nullptr, T_, E_, IN_);
  k_bf_t   <<<2048, 256, 0, stream>>>(b_bf, bT_bf, T_, E_);

  // ---- flash attention + gating (u_mod) ----
  k_flash<<<T_ / 16, 128, 0, stream>>>(u_bf, a_bf, bT_bf, um_bf);

  // ---- LSTM input-gate precompute: G = u_mod @ Wih^T + (b_ih + b_hh) ----
  dim3 ggate(G4_ / 256, T_ / 16);
  k_gemm_nt<<<ggate, 128, 0, stream>>>(um_bf, Wihf, Gfx, nullptr, biasf, T_, G4_, E_);
  k_gemm_nt<<<ggate, 128, 0, stream>>>(um_bf, Wihb, Gbx, nullptr, biasb, T_, G4_, E_);

  // ---- sequential recurrence, both directions concurrently ----
  k_lstm<<<2, 1024, 0, stream>>>(Gfx, Gbx, Whhf, Whhb, hf, hb);

  // ---- final linear ----
  k_final<<<T_, 128, 0, stream>>>(hf, hb, fin_w, fin_b, out);
}
